// SASRec_27754078667280
// MI455X (gfx1250) — matrix operand order repaired and checked by reading, not compiled, not run on hardware
//
#include <hip/hip_runtime.h>
#include <cstddef>

#define BB      2048
#define LSEQ    200
#define HDIM    128
#define NHEAD   2
#define NBLK    2
#define HDHEAD  64
#define QTILES  13                  // ceil(200/16)
#define NROW    (BB * LSEQ)         // 409600
#define SQRT_H  11.313708498984761f
#define FLOATMIN (-1000.0f)
#define LN_EPS  1e-8f

typedef __attribute__((ext_vector_type(16))) __bf16 v16bf;
typedef __attribute__((ext_vector_type(8)))  float  v8f;

__device__ __forceinline__ v8f wmma_bf16(v16bf a, v16bf b, v8f c) {
  // D = A(16x32 bf16) * B(32x16 bf16) + C(16x16 f32)
  return __builtin_amdgcn_wmma_f32_16x16x32_bf16(false, a, false, b, (short)0, c,
                                                 false, false);
}

__device__ __forceinline__ v8f vzero8() {
  v8f z;
#pragma unroll
  for (int i = 0; i < 8; ++i) z[i] = 0.0f;
  return z;
}

// Shared WMMA mainloop: one wave computes a 16-row M-tile x all 128 cols.
// acc[nt] = D tile for cols [nt*16, nt*16+16), rows rowBase..rowBase+15.
__device__ __forceinline__ void gemm_mainloop(const float* __restrict__ X,
                                              const __bf16* __restrict__ W,
                                              size_t rowA, int hi, int n,
                                              v8f acc[8]) {
#pragma unroll
  for (int ks = 0; ks < 4; ++ks) {
    // A fragment (16-bit A layout): e<8 -> K=e, e>=8 -> K=e+8; lanes>=16 add 8
    const float* pa = X + rowA * HDIM + ks * 32 + hi * 8;
    v16bf a;
#pragma unroll
    for (int e = 0; e < 8; ++e)  a[e] = (__bf16)pa[e];
#pragma unroll
    for (int e = 8; e < 16; ++e) a[e] = (__bf16)pa[e + 8];
#pragma unroll
    for (int nt = 0; nt < 8; ++nt) {
      // B[k][n] = W[nt*16+n][k]: lane n streams one bf16 weight row
      const __bf16* pb = W + (size_t)(nt * 16 + n) * HDIM + ks * 32 + hi * 8;
      v16bf b;
#pragma unroll
      for (int e = 0; e < 8; ++e)  b[e] = pb[e];
#pragma unroll
      for (int e = 8; e < 16; ++e) b[e] = pb[e + 8];
      acc[nt] = wmma_bf16(a, b, acc[nt]);
    }
  }
}

// ---------------------------------------------------------------------------
// Y = act(X @ W^T + bias)
// ---------------------------------------------------------------------------
__global__ __launch_bounds__(256) void gemm_wmma(const float* __restrict__ X,
                                                 const __bf16* __restrict__ W,
                                                 const float* __restrict__ bias,
                                                 float* __restrict__ Y, int relu) {
  const int lane = threadIdx.x & 31;
  const int widx = threadIdx.x >> 5;
  const int mt   = blockIdx.x * 8 + widx;
  const int hi   = (lane >> 4) & 1;
  const int n    = lane & 15;
  const size_t rowA = (size_t)mt * 16 + n;

  v8f acc[8];
#pragma unroll
  for (int t = 0; t < 8; ++t) acc[t] = vzero8();
  gemm_mainloop(X, W, rowA, hi, n, acc);

  const size_t rowBase = (size_t)mt * 16;
#pragma unroll
  for (int nt = 0; nt < 8; ++nt) {
    const float bv = bias[nt * 16 + n];
#pragma unroll
    for (int r = 0; r < 8; ++r) {
      float v = acc[nt][r] + bv;
      if (relu) v = fmaxf(v, 0.0f);
      Y[(rowBase + (size_t)(r + hi * 8)) * HDIM + nt * 16 + n] = v;
    }
  }
}

// ---------------------------------------------------------------------------
// Yo = LayerNorm(Res + (X @ W^T + bias)) * g + beta    (fused epilogue)
// One wave holds each output row fully in registers (8 vals/lane x 16 lanes),
// so the row mean/var are an 8-term in-register sum + 4 shfl_xor steps.
// ---------------------------------------------------------------------------
__global__ __launch_bounds__(256) void gemm_wmma_ln(const float* __restrict__ X,
                                                    const __bf16* __restrict__ W,
                                                    const float* __restrict__ bias,
                                                    const float* __restrict__ Res,
                                                    const float* __restrict__ g,
                                                    const float* __restrict__ beta,
                                                    float* __restrict__ Yo) {
  const int lane = threadIdx.x & 31;
  const int widx = threadIdx.x >> 5;
  const int mt   = blockIdx.x * 8 + widx;
  const int hi   = (lane >> 4) & 1;
  const int n    = lane & 15;
  const size_t rowA = (size_t)mt * 16 + n;

  v8f acc[8];
#pragma unroll
  for (int t = 0; t < 8; ++t) acc[t] = vzero8();
  gemm_mainloop(X, W, rowA, hi, n, acc);

  const size_t rowBase = (size_t)mt * 16;
  // bias + residual
#pragma unroll
  for (int nt = 0; nt < 8; ++nt) {
    const float bv = bias[nt * 16 + n];
#pragma unroll
    for (int r = 0; r < 8; ++r)
      acc[nt][r] += bv + Res[(rowBase + (size_t)(r + hi * 8)) * HDIM + nt * 16 + n];
  }
  // per-column LN params for this lane
  float gv[8], bb[8];
#pragma unroll
  for (int nt = 0; nt < 8; ++nt) {
    gv[nt] = g[nt * 16 + n];
    bb[nt] = beta[nt * 16 + n];
  }
  // per-row LayerNorm
#pragma unroll
  for (int r = 0; r < 8; ++r) {
    float s = 0.0f;
#pragma unroll
    for (int nt = 0; nt < 8; ++nt) s += acc[nt][r];
    s += __shfl_xor(s, 1); s += __shfl_xor(s, 2);
    s += __shfl_xor(s, 4); s += __shfl_xor(s, 8);
    const float mean = s * (1.0f / HDIM);
    float ss = 0.0f;
#pragma unroll
    for (int nt = 0; nt < 8; ++nt) {
      const float d = acc[nt][r] - mean;
      acc[nt][r] = d;
      ss += d * d;
    }
    ss += __shfl_xor(ss, 1); ss += __shfl_xor(ss, 2);
    ss += __shfl_xor(ss, 4); ss += __shfl_xor(ss, 8);
    const float inv = rsqrtf(ss * (1.0f / HDIM) + LN_EPS);
    float* po = Yo + (rowBase + (size_t)(r + hi * 8)) * HDIM;
#pragma unroll
    for (int nt = 0; nt < 8; ++nt)
      po[nt * 16 + n] = acc[nt][r] * inv * gv[nt] + bb[nt];
  }
}

// ---------------------------------------------------------------------------
// Flash attention: one wave per (batch, head, 16-row q-tile). hd=64, L=200.
// ---------------------------------------------------------------------------
__global__ __launch_bounds__(256) void attn_flash(const float* __restrict__ Q,
                                                  const float* __restrict__ K,
                                                  const float* __restrict__ V,
                                                  float* __restrict__ O) {
  __shared__ __bf16 ldsP[8][16 * 32];  // 1KB per wave
  const int lane = threadIdx.x & 31;
  const int widx = threadIdx.x >> 5;
  const int gw   = blockIdx.x * 8 + widx;
  const int qt = gw % QTILES;
  const int h  = (gw / QTILES) % NHEAD;
  const int b  = gw / (QTILES * NHEAD);
  if (b >= BB) return;  // wave-uniform
  const int hi = (lane >> 4) & 1;
  const int n  = lane & 15;
  const int qbase = qt * 16;

  const size_t base = (size_t)b * LSEQ * HDIM + (size_t)h * HDHEAD;
  const float* Qb = Q + base;
  const float* Kb = K + base;
  const float* Vb = V + base;
  float*       Ob = O + base;

  int qrow = qbase + n; if (qrow > LSEQ - 1) qrow = LSEQ - 1;
  v16bf aq[2];
#pragma unroll
  for (int ds = 0; ds < 2; ++ds) {
    const float* p = Qb + (size_t)qrow * HDIM + ds * 32 + hi * 16;
    v16bf a;
#pragma unroll
    for (int e = 0; e < 8; ++e)  a[e] = (__bf16)p[e];
#pragma unroll
    for (int e = 8; e < 16; ++e) a[e] = (__bf16)p[e];
    aq[ds] = a;
  }

  v8f acc[4];
#pragma unroll
  for (int t = 0; t < 4; ++t) acc[t] = vzero8();
  float mrow[8], lrow[8];
#pragma unroll
  for (int r = 0; r < 8; ++r) { mrow[r] = -1e30f; lrow[r] = 0.0f; }

  const int qmaxv  = (qbase + 15 < LSEQ - 1) ? (qbase + 15) : (LSEQ - 1);
  const int nstrip = qmaxv / 32 + 1;

  for (int st = 0; st < nstrip; ++st) {
    const int kk = st * 32;
    v8f s0 = vzero8(), s1 = vzero8();
#pragma unroll
    for (int ds = 0; ds < 2; ++ds) {
      int kr0 = kk + n;      if (kr0 > LSEQ - 1) kr0 = LSEQ - 1;
      int kr1 = kk + 16 + n; if (kr1 > LSEQ - 1) kr1 = LSEQ - 1;
      const float* p0 = Kb + (size_t)kr0 * HDIM + ds * 32 + hi * 16;
      const float* p1 = Kb + (size_t)kr1 * HDIM + ds * 32 + hi * 16;
      v16bf b0, b1;
#pragma unroll
      for (int e = 0; e < 16; ++e) { b0[e] = (__bf16)p0[e]; b1[e] = (__bf16)p1[e]; }
      s0 = wmma_bf16(aq[ds], b0, s0);
      s1 = wmma_bf16(aq[ds], b1, s1);
    }
#pragma unroll
    for (int r = 0; r < 8; ++r) {
      const int row = qbase + r + hi * 8;
      const int c0 = kk + n, c1 = kk + 16 + n;
      float v0 = s0[r] * 0.125f;
      float v1 = s1[r] * 0.125f;
      if (c0 > row || c0 >= LSEQ) v0 = FLOATMIN;
      if (c1 > row || c1 >= LSEQ) v1 = FLOATMIN;
      float tv = fmaxf(v0, v1);
      tv = fmaxf(tv, __shfl_xor(tv, 1));
      tv = fmaxf(tv, __shfl_xor(tv, 2));
      tv = fmaxf(tv, __shfl_xor(tv, 4));
      tv = fmaxf(tv, __shfl_xor(tv, 8));
      const float mn    = fmaxf(mrow[r], tv);
      const float alpha = __expf(mrow[r] - mn);
      mrow[r] = mn;
      const float p0 = __expf(v0 - mn);
      const float p1 = __expf(v1 - mn);
      s0[r] = p0; s1[r] = p1;
      float rs = p0 + p1;
      rs += __shfl_xor(rs, 1);
      rs += __shfl_xor(rs, 2);
      rs += __shfl_xor(rs, 4);
      rs += __shfl_xor(rs, 8);
      lrow[r] = lrow[r] * alpha + rs;
#pragma unroll
      for (int dt = 0; dt < 4; ++dt) acc[dt][r] *= alpha;
    }
    // stage P through LDS: D-layout -> A-layout, bf16 (wave-local ordering)
    __bf16* lp = &ldsP[widx][0];
#pragma unroll
    for (int r = 0; r < 8; ++r) {
      const int row = r + hi * 8;
      lp[row * 32 + n]      = (__bf16)s0[r];
      lp[row * 32 + 16 + n] = (__bf16)s1[r];
    }
    asm volatile("s_wait_dscnt 0" ::: "memory");
    v16bf ap;
    {
      const __bf16* p = lp + n * 32 + hi * 16;
#pragma unroll
      for (int e = 0; e < 8; ++e)  ap[e] = p[e];
#pragma unroll
      for (int e = 8; e < 16; ++e) ap[e] = p[e];
    }
#pragma unroll
    for (int dt = 0; dt < 4; ++dt) {
      v16bf bv;
#pragma unroll
      for (int e = 0; e < 16; ++e) {
        int kr = kk + e + hi * 16;
        if (kr > LSEQ - 1) kr = LSEQ - 1;  // masked rows have P==0
        bv[e] = (__bf16)Vb[(size_t)kr * HDIM + dt * 16 + n];
      }
      acc[dt] = wmma_bf16(ap, bv, acc[dt]);
    }
  }

#pragma unroll
  for (int r = 0; r < 8; ++r) {
    const int row = qbase + r + hi * 8;
    if (row < LSEQ) {
      const float rinv = 1.0f / lrow[r];
      float* po = Ob + (size_t)row * HDIM;
#pragma unroll
      for (int dt = 0; dt < 4; ++dt)
        po[dt * 16 + n] = acc[dt][r] * rinv;
    }
  }
}

// ---------------------------------------------------------------------------
// X[b,l,:] = item_table[log] * sqrt(H) + pos_table[(l+1)*(log!=0)]
// ---------------------------------------------------------------------------
__global__ __launch_bounds__(256) void embed_k(const int* __restrict__ logs,
                                               const float* __restrict__ itab,
                                               const float* __restrict__ ptab,
                                               float* __restrict__ X) {
  const size_t t = (size_t)blockIdx.x * blockDim.x + threadIdx.x;
  const size_t row = t >> 5;
  const int sub = (int)(t & 31);
  const int it = logs[row];
  const int l = (int)(row % LSEQ);
  const int pidx = (it != 0) ? (l + 1) : 0;
  const float4 iv = *(const float4*)(itab + (size_t)it * HDIM + sub * 4);
  const float4 pv = *(const float4*)(ptab + (size_t)pidx * HDIM + sub * 4);
  float4 o;
  o.x = iv.x * SQRT_H + pv.x;
  o.y = iv.y * SQRT_H + pv.y;
  o.z = iv.z * SQRT_H + pv.z;
  o.w = iv.w * SQRT_H + pv.w;
  *(float4*)(X + row * HDIM + sub * 4) = o;
}

// ---------------------------------------------------------------------------
// Fused final LayerNorm + pos/neg logits. One wave per token row.
// ---------------------------------------------------------------------------
__global__ __launch_bounds__(256) void logits_ln_k(const float* __restrict__ X,
                                                   const float* __restrict__ g,
                                                   const float* __restrict__ beta,
                                                   const float* __restrict__ itab,
                                                   const int* __restrict__ pos,
                                                   const int* __restrict__ neg,
                                                   float* __restrict__ out) {
  const int lane = threadIdx.x & 31;
  const int widx = threadIdx.x >> 5;
  const size_t idx = (size_t)blockIdx.x * 8 + widx;
  const size_t off = idx * HDIM + lane * 4;
  const float4 x = *(const float4*)(X + off);
  // row mean / variance (128 elems = 32 lanes x 4)
  float s = x.x + x.y + x.z + x.w;
  s += __shfl_xor(s, 1); s += __shfl_xor(s, 2); s += __shfl_xor(s, 4);
  s += __shfl_xor(s, 8); s += __shfl_xor(s, 16);
  const float mean = s * (1.0f / HDIM);
  float4 d;
  d.x = x.x - mean; d.y = x.y - mean; d.z = x.z - mean; d.w = x.w - mean;
  float ss = d.x * d.x + d.y * d.y + d.z * d.z + d.w * d.w;
  ss += __shfl_xor(ss, 1); ss += __shfl_xor(ss, 2); ss += __shfl_xor(ss, 4);
  ss += __shfl_xor(ss, 8); ss += __shfl_xor(ss, 16);
  const float inv = rsqrtf(ss * (1.0f / HDIM) + LN_EPS);
  const float4 gg = *(const float4*)(g + lane * 4);
  const float4 bb = *(const float4*)(beta + lane * 4);
  float4 f;
  f.x = d.x * inv * gg.x + bb.x;
  f.y = d.y * inv * gg.y + bb.y;
  f.z = d.z * inv * gg.z + bb.z;
  f.w = d.w * inv * gg.w + bb.w;
  // dot with pos/neg item embeddings
  const int pi = pos[idx];
  const int ni = neg[idx];
  const float4 pv = *(const float4*)(itab + (size_t)pi * HDIM + lane * 4);
  const float4 nv = *(const float4*)(itab + (size_t)ni * HDIM + lane * 4);
  float dp = f.x * pv.x + f.y * pv.y + f.z * pv.z + f.w * pv.w;
  float dn = f.x * nv.x + f.y * nv.y + f.z * nv.z + f.w * nv.w;
  dp += __shfl_xor(dp, 1); dp += __shfl_xor(dp, 2); dp += __shfl_xor(dp, 4);
  dp += __shfl_xor(dp, 8); dp += __shfl_xor(dp, 16);
  dn += __shfl_xor(dn, 1); dn += __shfl_xor(dn, 2); dn += __shfl_xor(dn, 4);
  dn += __shfl_xor(dn, 8); dn += __shfl_xor(dn, 16);
  if (lane == 0) {
    out[idx] = dp;
    out[(size_t)NROW + idx] = dn;
  }
}

__global__ __launch_bounds__(256) void cvt_bf16(const float* __restrict__ src,
                                                __bf16* __restrict__ dst, int nElem) {
  const int t = blockIdx.x * blockDim.x + threadIdx.x;
  if (t < nElem) dst[t] = (__bf16)src[t];
}

// ---------------------------------------------------------------------------
extern "C" void kernel_launch(void* const* d_in, const int* in_sizes, int n_in,
                              void* d_out, int out_size, void* d_ws, size_t ws_size,
                              hipStream_t stream) {
  (void)in_sizes; (void)n_in; (void)out_size; (void)ws_size;
  const int*   logs = (const int*)d_in[1];
  const int*   pos  = (const int*)d_in[2];
  const int*   neg  = (const int*)d_in[3];
  const float* itab = (const float*)d_in[4];
  const float* ptab = (const float*)d_in[5];
  const float* qw = (const float*)d_in[6];
  const float* kw = (const float*)d_in[7];
  const float* vw = (const float*)d_in[8];
  const float* ow = (const float*)d_in[9];
  const float* qb = (const float*)d_in[10];
  const float* kb = (const float*)d_in[11];
  const float* vb = (const float*)d_in[12];
  const float* ob = (const float*)d_in[13];
  const float* w1 = (const float*)d_in[14];
  const float* b1 = (const float*)d_in[15];
  const float* w2 = (const float*)d_in[16];
  const float* b2 = (const float*)d_in[17];
  const float* ln1g = (const float*)d_in[18];
  const float* ln1b = (const float*)d_in[19];
  const float* ln2g = (const float*)d_in[20];
  const float* ln2b = (const float*)d_in[21];
  const float* lnfg = (const float*)d_in[22];
  const float* lnfb = (const float*)d_in[23];

  const size_t SZ = (size_t)NROW * HDIM;  // 52,428,800 floats per buffer
  float* X  = (float*)d_ws;
  float* A1 = X + SZ;
  float* A2 = A1 + SZ;
  float* A3 = A2 + SZ;
  float* A4 = A3 + SZ;
  __bf16* WB = (__bf16*)(A4 + SZ);
  const int WSZ = NBLK * HDIM * HDIM;  // 32768 per weight tensor
  __bf16* qwb = WB;
  __bf16* kwb = qwb + WSZ;
  __bf16* vwb = kwb + WSZ;
  __bf16* owb = vwb + WSZ;
  __bf16* w1b = owb + WSZ;
  __bf16* w2b = w1b + WSZ;

  cvt_bf16<<<WSZ / 256, 256, 0, stream>>>(qw, qwb, WSZ);
  cvt_bf16<<<WSZ / 256, 256, 0, stream>>>(kw, kwb, WSZ);
  cvt_bf16<<<WSZ / 256, 256, 0, stream>>>(vw, vwb, WSZ);
  cvt_bf16<<<WSZ / 256, 256, 0, stream>>>(ow, owb, WSZ);
  cvt_bf16<<<WSZ / 256, 256, 0, stream>>>(w1, w1b, WSZ);
  cvt_bf16<<<WSZ / 256, 256, 0, stream>>>(w2, w2b, WSZ);

  embed_k<<<NROW * 32 / 256, 256, 0, stream>>>(logs, itab, ptab, X);

  const int GEMM_BLOCKS = NROW / 16 / 8;           // 3200
  const int ATTN_BLOCKS = BB * NHEAD * QTILES / 8; // 6656
  const int ROW_BLOCKS  = NROW / 8;                // 51200
  const int HH = HDIM * HDIM;

  for (int i = 0; i < NBLK; ++i) {
    gemm_wmma<<<GEMM_BLOCKS, 256, 0, stream>>>(X, qwb + (size_t)i * HH, qb + i * HDIM, A1, 0);
    gemm_wmma<<<GEMM_BLOCKS, 256, 0, stream>>>(X, kwb + (size_t)i * HH, kb + i * HDIM, A2, 0);
    gemm_wmma<<<GEMM_BLOCKS, 256, 0, stream>>>(X, vwb + (size_t)i * HH, vb + i * HDIM, A3, 0);
    attn_flash<<<ATTN_BLOCKS, 256, 0, stream>>>(A1, A2, A3, A4);
    // O-projection + residual(X) + LN1, written back to X (row-disjoint per wave)
    gemm_wmma_ln<<<GEMM_BLOCKS, 256, 0, stream>>>(A4, owb + (size_t)i * HH, ob + i * HDIM,
                                                  X, ln1g + i * HDIM, ln1b + i * HDIM, X);
    gemm_wmma<<<GEMM_BLOCKS, 256, 0, stream>>>(X, w1b + (size_t)i * HH, b1 + i * HDIM, A2, 1);
    // FFN-2 + residual(X) + LN2, written back to X
    gemm_wmma_ln<<<GEMM_BLOCKS, 256, 0, stream>>>(A2, w2b + (size_t)i * HH, b2 + i * HDIM,
                                                  X, ln2g + i * HDIM, ln2b + i * HDIM, X);
  }
  // fused final LN + pos/neg logits
  logits_ln_k<<<ROW_BLOCKS, 256, 0, stream>>>(X, lnfg, lnfb, itab, pos, neg, (float*)d_out);
}
